// Fusion2line_31447750541867
// MI455X (gfx1250) — matrix operand order repaired and checked by reading, not compiled, not run on hardware
//
#include <hip/hip_runtime.h>
#include <hip/hip_bf16.h>
#include <stdint.h>

// Problem constants (match reference)
#define NCV   120000
#define NFV   (2*NCV)
#define CINV  128
#define COUTV 64

typedef __attribute__((ext_vector_type(16))) __bf16 v16bf;
typedef __attribute__((ext_vector_type(8)))  __bf16 v8bf;
typedef __attribute__((ext_vector_type(8)))  float  v8f;

static __device__ __forceinline__ __bf16 f2bf(float f) {
  unsigned u = __builtin_bit_cast(unsigned, f);
  u += 0x7FFFu + ((u >> 16) & 1u);          // round-to-nearest-even
  unsigned short h = (unsigned short)(u >> 16);
  return __builtin_bit_cast(__bf16, h);
}

// Native device-scope FP32 atomic add (no return) in GVS addressing mode:
// uniform SGPR base + per-lane 32-bit byte offset. Avoids both the CAS-loop
// fallback of atomicAdd(float*) and per-element 64-bit VGPR address math.
// Tracked by STOREcnt; drained by the implicit S_WAIT_IDLE at s_endpgm.
static __device__ __forceinline__ void gfadd_off(const float* base, int offB,
                                                 float v) {
  asm volatile("global_atomic_add_f32 %0, %1, %2 scope:SCOPE_DEV"
               :: "v"(offB), "v"(v), "s"(base) : "memory");
}

// ---------------------------------------------------------------------------
// Weight prep: f32 [k][K][64] -> bf16 WMMA B-fragments.
// Fragment order: (((k*4 + nblk)*KB + kblk)*32 + lane)*16 + e
// Lane layout (16-bit B, 32x16 per WMMA): lanes 0-15 hold K=kblk*32+{0..7,16..23} at
// col n=lane; lanes 16-31 hold K=kblk*32+{8..15,24..31} at col n=lane-16.
// ---------------------------------------------------------------------------
__global__ void prep_frag_kernel(const float* __restrict__ W,
                                 __bf16* __restrict__ frag,
                                 int total, int KB) {
  int i = blockIdx.x * blockDim.x + threadIdx.x;
  if (i >= total) return;
  int e    = i & 15;
  int lane = (i >> 4) & 31;
  int q    = i >> 9;
  int kblk = q % KB;  q /= KB;
  int nblk = q & 3;
  int k    = q >> 2;
  int Kdim = KB * 32;
  int n = nblk * 16 + (lane & 15);
  int K = kblk * 32 + ((lane & 16) ? 8 : 0) + (e & 7) + ((e & 8) ? 16 : 0);
  frag[i] = f2bf(W[((size_t)k * Kdim + K) * COUTV + n]);
}

// ---------------------------------------------------------------------------
// x = feats_x + feats_skip -> bf16, plus one zero pad row (row NC)
// ---------------------------------------------------------------------------
__global__ void fuse_kernel(const float* __restrict__ a,
                            const float* __restrict__ b,
                            __bf16* __restrict__ o) {
  int i = blockIdx.x * blockDim.x + threadIdx.x;
  const int total = (NCV + 1) * CINV;
  if (i >= total) return;
  float v = (i < NCV * CINV) ? (a[i] + b[i]) : 0.f;
  o[i] = f2bf(v);
}

// ---------------------------------------------------------------------------
// Sparse conv: gather (bf16 rows) -> WMMA bf16 GEMM -> scatter atomic f32.
// grid.y = kernel offset k; one wave = one 16-row pair tile -> 16x64 output.
// ---------------------------------------------------------------------------
template<int KB>
__global__ __launch_bounds__(256)
void spconv_wmma_kernel(const __bf16* __restrict__ feats,
                        const __bf16* __restrict__ frag,
                        const int*    __restrict__ pairs,
                        float*        __restrict__ out,
                        int nOutGuard) {
  const int ntiles = NCV / 16;                     // 7500
  int tile = blockIdx.x * 8 + (threadIdx.x >> 5);
  if (tile >= ntiles) return;                      // wave-uniform exit
  int lane = threadIdx.x & 31;
  int k    = blockIdx.y;
  int p0   = tile * 16;
  const int* pk = pairs + (size_t)k * NCV * 2;
  const int Kdim = KB * 32;

  // -------- gather A fragments (ISA 16-bit A 16x32 layout) ----------------
  int iidx = pk[(p0 + (lane & 15)) * 2 + 0];       // padded entries -> row NC (zeros)
  const __bf16* arow = feats + (size_t)iidx * Kdim;
  int half = (lane & 16) ? 8 : 0;
  v16bf A[KB];
#pragma unroll
  for (int kb = 0; kb < KB; ++kb) {
    v8bf lo = *(const v8bf*)(arow + kb * 32 + half);
    v8bf hi = *(const v8bf*)(arow + kb * 32 + half + 16);
    A[kb] = __builtin_shufflevector(lo, hi, 0,1,2,3,4,5,6,7,8,9,10,11,12,13,14,15);
  }

  // -------- GEMM: 4 N-tiles x KB K-blocks of v_wmma_f32_16x16x32_bf16 -----
  const v16bf* bfr = (const v16bf*)frag + (size_t)k * 4 * KB * 32 + lane;
  v8f acc[4];
#pragma unroll
  for (int nb = 0; nb < 4; ++nb) {
    v8f c = {};
#pragma unroll
    for (int kb = 0; kb < KB; ++kb) {
      v16bf B = bfr[((size_t)nb * KB + kb) * 32];
      c = __builtin_amdgcn_wmma_f32_16x16x32_bf16(false, A[kb], false, B,
                                                  (short)0, c, false, false);
    }
    acc[nb] = c;
  }

  // -------- scatter-add (C layout: lane<16 -> rows 0..7, lane>=16 -> 8..15)
  int orow[8];
#pragma unroll
  for (int v = 0; v < 8; ++v) {
    int m = (lane & 16) ? (8 + v) : v;
    orow[v] = pk[(p0 + m) * 2 + 1];
  }
  int col = lane & 15;
#pragma unroll
  for (int nb = 0; nb < 4; ++nb) {
    int cbase = (nb * 16 + col) * 4;               // byte offset within row
#pragma unroll
    for (int v = 0; v < 8; ++v) {
      int o = orow[v];
      if (o < nOutGuard)
        gfadd_off(out, (o << 8) + cbase, acc[nb][v]);  // row stride 64*4=256B
    }
  }
}

// ---------------------------------------------------------------------------
// BN batch stats: per-channel sum & sum-of-squares (optional leaky pre-act)
// ---------------------------------------------------------------------------
__global__ void stats_kernel(const float* __restrict__ buf,
                             float* __restrict__ stats, int leaky) {
  __shared__ float ssum[256];
  __shared__ float ssq[256];
  int c  = threadIdx.x & 63;
  int rg = threadIdx.x >> 6;
  float s = 0.f, q = 0.f;
  for (int r = blockIdx.x * 4 + rg; r < NCV; r += gridDim.x * 4) {
    float v = buf[(size_t)r * 64 + c];
    if (leaky) v = v > 0.f ? v : 0.01f * v;
    s += v; q += v * v;
  }
  ssum[threadIdx.x] = s; ssq[threadIdx.x] = q;
  __syncthreads();
  if (threadIdx.x < 64) {
    s = ssum[c] + ssum[64 + c] + ssum[128 + c] + ssum[192 + c];
    q = ssq[c]  + ssq[64 + c]  + ssq[128 + c]  + ssq[192 + c];
    gfadd_off(stats, c * 4, s);
    gfadd_off(stats, (64 + c) * 4, q);
  }
}

// stats -> affine coefficients: y_bn = y*s + t
__global__ void finalize_kernel(const float* __restrict__ st,
                                const float* __restrict__ g,
                                const float* __restrict__ b,
                                float* __restrict__ coef) {
  int c = threadIdx.x;
  if (c < 64) {
    float m  = st[c] * (1.f / NCV);
    float vr = st[64 + c] * (1.f / NCV) - m * m;
    float rs = rsqrtf(vr + 1e-5f);
    float s  = g[c] * rs;
    coef[c] = s;
    coef[64 + c] = b[c] - m * s;
  }
}

// upA_bn = bf16(affine(leaky(raw))), zero pad row
__global__ void apply_bn_leaky_kernel(const float* __restrict__ raw,
                                      const float* __restrict__ coef,
                                      __bf16* __restrict__ out) {
  int i = blockIdx.x * blockDim.x + threadIdx.x;
  const int total = (NCV + 1) * 64;
  if (i >= total) return;
  int r = i >> 6, c = i & 63;
  float v = 0.f;
  if (r < NCV) {
    float x = raw[i];
    x = x > 0.f ? x : 0.01f * x;
    v = x * coef[c] + coef[64 + c];
  }
  out[i] = f2bf(v);
}

// z = bn1(E1) + bn2(E2) -> bf16, zero pad row
__global__ void combine_kernel(const float* __restrict__ e1,
                               const float* __restrict__ e2,
                               const float* __restrict__ c1,
                               const float* __restrict__ c2,
                               __bf16* __restrict__ out) {
  int i = blockIdx.x * blockDim.x + threadIdx.x;
  const int total = (NCV + 1) * 64;
  if (i >= total) return;
  int r = i >> 6, c = i & 63;
  float v = 0.f;
  if (r < NCV)
    v = (e1[i] * c1[c] + c1[64 + c]) + (e2[i] * c2[c] + c2[64 + c]);
  out[i] = f2bf(v);
}

// ---------------------------------------------------------------------------
extern "C" void kernel_launch(void* const* d_in, const int* in_sizes, int n_in,
                              void* d_out, int out_size, void* d_ws, size_t ws_size,
                              hipStream_t stream) {
  (void)in_sizes; (void)n_in; (void)out_size; (void)ws_size;
  const float* feats_x    = (const float*)d_in[0];
  const float* feats_skip = (const float*)d_in[1];
  const float* W_trans    = (const float*)d_in[2];
  const float* W1         = (const float*)d_in[3];
  const float* W2         = (const float*)d_in[4];
  const float* W_up       = (const float*)d_in[5];
  const float* g0 = (const float*)d_in[6];
  const float* b0 = (const float*)d_in[7];
  const float* g1 = (const float*)d_in[8];
  const float* b1 = (const float*)d_in[9];
  const float* g2 = (const float*)d_in[10];
  const float* b2 = (const float*)d_in[11];
  const int* p33 = (const int*)d_in[12];
  const int* p13 = (const int*)d_in[13];
  const int* p31 = (const int*)d_in[14];
  const int* pup = (const int*)d_in[15];
  float* outF = (float*)d_out;

  // -------- workspace bump allocator (with lifetime aliasing) --------------
  char* w = (char*)d_ws;
  size_t off = 0;
  auto alloc = [&](size_t bytes) -> void* {
    void* p = (void*)(w + off);
    off += (bytes + 255) & ~(size_t)255;
    return p;
  };
  __bf16* fpx   = (__bf16*)alloc((size_t)(NCV + 1) * CINV * 2);   // 30.7MB
  float*  upA   = (float*) alloc((size_t)(NCV + 1) * COUTV * 4);  // 30.7MB
  float*  E2    = (float*) alloc((size_t)(NCV + 1) * COUTV * 4);  // 30.7MB
  __bf16* upAbn = (__bf16*)alloc((size_t)(NCV + 1) * COUTV * 2);  // 15.4MB
  __bf16* fragT = (__bf16*)alloc((size_t)27 * 4 * 4 * 512 * 2);
  __bf16* frag1 = (__bf16*)alloc((size_t)9  * 4 * 2 * 512 * 2);
  __bf16* frag2 = (__bf16*)alloc((size_t)9  * 4 * 2 * 512 * 2);
  __bf16* fragU = (__bf16*)alloc((size_t)27 * 4 * 2 * 512 * 2);
  float*  stats = (float*) alloc(1024 * 4);
  // aliases (stream-ordered lifetimes: fpx dead after spconv1, upA after apply)
  float*  E1   = (float*)fpx;
  __bf16* zbuf = (__bf16*)upA;

  const int TPB = 256;
  const int TBLK = (NCV / 16 + 7) / 8;   // x-blocks per kernel offset (938)
  // -------- weight fragments ----------------------------------------------
  {
    int tT = 27 * 4 * 4 * 512, t9 = 9 * 4 * 2 * 512, tU = 27 * 4 * 2 * 512;
    prep_frag_kernel<<<(tT + TPB - 1) / TPB, TPB, 0, stream>>>(W_trans, fragT, tT, 4);
    prep_frag_kernel<<<(t9 + TPB - 1) / TPB, TPB, 0, stream>>>(W1, frag1, t9, 2);
    prep_frag_kernel<<<(t9 + TPB - 1) / TPB, TPB, 0, stream>>>(W2, frag2, t9, 2);
    prep_frag_kernel<<<(tU + TPB - 1) / TPB, TPB, 0, stream>>>(W_up, fragU, tU, 2);
  }
  // -------- fuse residual -> bf16 -----------------------------------------
  {
    int total = (NCV + 1) * CINV;
    fuse_kernel<<<(total + TPB - 1) / TPB, TPB, 0, stream>>>(feats_x, feats_skip, fpx);
  }
  // -------- layer 1: 3x3x3 subm, 128 -> 64 --------------------------------
  hipMemsetAsync(upA, 0, (size_t)(NCV + 1) * COUTV * 4, stream);
  hipMemsetAsync(stats, 0, 1024 * 4, stream);
  spconv_wmma_kernel<4><<<dim3(TBLK, 27), TPB, 0, stream>>>(
      fpx, fragT, p33, upA, NCV + 1);
  // -------- leaky + BN0 stats, fold into affine ---------------------------
  stats_kernel<<<480, TPB, 0, stream>>>(upA, stats + 0, 1);
  finalize_kernel<<<1, 64, 0, stream>>>(stats + 0, g0, b0, stats + 384);
  {
    int total = (NCV + 1) * COUTV;
    apply_bn_leaky_kernel<<<(total + TPB - 1) / TPB, TPB, 0, stream>>>(
        upA, stats + 384, upAbn);
  }
  // -------- layers 2/3: 1x3x3 and 3x1x3 subm, 64 -> 64 --------------------
  hipMemsetAsync(E1, 0, (size_t)(NCV + 1) * COUTV * 4, stream);  // aliases fpx (dead)
  hipMemsetAsync(E2, 0, (size_t)(NCV + 1) * COUTV * 4, stream);
  spconv_wmma_kernel<2><<<dim3(TBLK, 9), TPB, 0, stream>>>(
      upAbn, frag1, p13, E1, NCV + 1);
  spconv_wmma_kernel<2><<<dim3(TBLK, 9), TPB, 0, stream>>>(
      upAbn, frag2, p31, E2, NCV + 1);
  // -------- BN1 + BN2 stats, combine z = bn1(E1)+bn2(E2) ------------------
  stats_kernel<<<480, TPB, 0, stream>>>(E1, stats + 128, 0);
  stats_kernel<<<480, TPB, 0, stream>>>(E2, stats + 256, 0);
  finalize_kernel<<<1, 64, 0, stream>>>(stats + 128, g1, b1, stats + 512);
  finalize_kernel<<<1, 64, 0, stream>>>(stats + 256, g2, b2, stats + 640);
  {
    int total = (NCV + 1) * COUTV;
    combine_kernel<<<(total + TPB - 1) / TPB, TPB, 0, stream>>>(
        E1, E2, stats + 512, stats + 640, zbuf);   // zbuf aliases upA (dead)
  }
  // -------- upsample: inverse 3x3x3, 64 -> 64, scatter into d_out ---------
  hipMemsetAsync(outF, 0, (size_t)NFV * COUTV * 4, stream);
  spconv_wmma_kernel<2><<<dim3(TBLK, 27), TPB, 0, stream>>>(
      zbuf, fragU, pup, outF, NFV);                // pad rows (idx==NF) dropped
}
